// Cluster_layer_6743098654913
// MI455X (gfx1250) — compile-verified, matrix operand order checked by reading, not run on hardware
//
#include <hip/hip_runtime.h>
#include <math.h>

// ---------------------------------------------------------------------------
// CDNA5 (gfx1250) Cluster_layer forward.
// All contractions via v_wmma_f32_16x16x32_f16 (f32 accum).
// GEMM operands pre-converted to f16 in global (halves HBM traffic),
// async global->LDS tile staging (ASYNCcnt) when available, b128 data path,
// next-slab prefetch (global_prefetch_b8).
// ---------------------------------------------------------------------------

typedef __attribute__((ext_vector_type(16))) _Float16 v16h;
typedef __attribute__((ext_vector_type(8)))  _Float16 v8h;
typedef __attribute__((ext_vector_type(2)))  _Float16 v2h;
typedef __attribute__((ext_vector_type(8)))  float    v8f;
typedef __attribute__((ext_vector_type(4)))  int      i32x4;

#define WMMA_F16(A_, B_, C_) \
  __builtin_amdgcn_wmma_f32_16x16x32_f16(false, (A_), false, (B_), (short)0, (C_), false, false)

// ---- async global -> LDS staging (gfx1250 GLOBAL_LOAD_ASYNC_TO_LDS_B128) ----
// Probe-confirmed: builtin exists, 4 args, param type = pointer to 4xi32.
#if __has_builtin(__builtin_amdgcn_global_load_async_to_lds_b128)
#define HAVE_ASYNC_LDS 1
__device__ __forceinline__ void async_copy_b128(const void* g, void* l) {
  __builtin_amdgcn_global_load_async_to_lds_b128((i32x4*)g, (i32x4*)l, 0, 0);
}
__device__ __forceinline__ void wait_async() {
#if __has_builtin(__builtin_amdgcn_s_wait_asynccnt)
  __builtin_amdgcn_s_wait_asynccnt(0);
#else
  asm volatile("s_wait_asynccnt 0x0" ::: "memory");
#endif
}
#else
#define HAVE_ASYNC_LDS 0
#endif

// Fragment fetch from an f16 LDS tile. Per ISA 7.12.2 the 16-bit A/B fragment
// for lane L is two contiguous 8-half chunks at k = kb and k = 16+kb
// (kb = 0 for lanes 0-15, 8 for lanes 16-31), row r = L&15.
__device__ __forceinline__ v16h frag_lds(const _Float16* base, int stride, int k0, int lane) {
  const int r = lane & 15;
  const int kb = (lane >> 4) ? 8 : 0;
  const _Float16* p = base + r * stride + k0 + kb;
  const v8h lo = *(const v8h*)p;
  const v8h hi = *(const v8h*)(p + 16);
  v16h o;
#pragma unroll
  for (int i = 0; i < 8; ++i) { o[i] = lo[i]; o[8 + i] = hi[i]; }
  return o;
}

// Fragment straight from f16 global (rows r0..r0+15, k0..k0+31), 2x b128.
__device__ __forceinline__ v16h frag_g16h(const _Float16* __restrict__ A, int lda,
                                          int r0, int k0, int lane) {
  const int r = r0 + (lane & 15);
  const int kb = (lane >> 4) ? 8 : 0;
  const _Float16* p = A + (long)r * lda + k0 + kb;
  const v8h lo = *(const v8h*)p;
  const v8h hi = *(const v8h*)(p + 16);
  v16h o;
#pragma unroll
  for (int i = 0; i < 8; ++i) { o[i] = lo[i]; o[8 + i] = hi[i]; }
  return o;
}

__device__ __forceinline__ void store_tile(float* __restrict__ Cb, _Float16* __restrict__ Ch,
                                           int ldc, int row0, int col0,
                                           int M, int Nn, v8f acc, float alpha, float beta,
                                           const float* __restrict__ bias, int lane) {
  const int n = col0 + (lane & 15);
  const int rb = row0 + ((lane >> 4) ? 8 : 0);
  if (n < Nn) {
    const float bv = bias ? bias[n] : 0.f;
#pragma unroll
    for (int j = 0; j < 8; ++j) {
      const int r = rb + j;
      if (r < M) {
        const long idx = (long)r * ldc + n;
        const float old = (beta != 0.f) ? Cb[idx] : 0.f;
        const float res = alpha * (acc[j] + bv) + beta * old;
        Cb[idx] = res;
        if (Ch) Ch[idx] = (_Float16)res;
      }
    }
  }
}

// Generic batched-strided GEMM: C = alpha*(A @ op(B) + bias) + beta*C
// A: (M,K) f16 rowmajor.  blayKN==0: B f16 (Nn,K);  blayKN==1: B f16 (K,Nn).
// K % 32 == 0, lda/ldb % 8 == 0. Block tile 64x128, wave tile 32x32.
// Staging: async b128 copies global->LDS (ASYNCcnt) when available, else
// b128 load/store pairs; fragments = 2x ds_load_b128.
// Out-of-range staging reads tolerated (callers guarantee readable slack).
__global__ __launch_bounds__(256)
void gemm_wmma_f16(const _Float16* __restrict__ A, long sA, int lda,
                   const _Float16* __restrict__ Bm, long sB, int ldb,
                   float* __restrict__ Cm, _Float16* __restrict__ Chm, long sC, int ldc,
                   int M, int Nn, int K, int blayKN,
                   float alpha, float beta, const float* __restrict__ bias) {
  __shared__ _Float16 Ah[64][40];   // 64 rows x 32 k (stride 40)
  __shared__ _Float16 Bh[128][40];  // 128 n  x 32 k (stride 40)
  const int t = threadIdx.x;
  const int lane = t & 31, wv = t >> 5;
  const int row0 = blockIdx.y * 64;
  const int col0 = blockIdx.x * 128;
  const _Float16* Ab = A + (long)blockIdx.z * sA;
  const _Float16* Bb = Bm + (long)blockIdx.z * sB;
  float* Cb = Cm + (long)blockIdx.z * sC;
  _Float16* Chb = Chm ? (Chm + (long)blockIdx.z * sC) : (_Float16*)0;

  const int wr = (wv >> 2) * 32;  // local row of wave tile
  const int wc = (wv & 3) * 32;   // local col of wave tile

  // staging coordinates
  const int ar = t >> 2, ak = (t & 3) * 8;         // A: row 0..63, 8 halves
  const int bn = t >> 1, bk = (t & 1) * 16;        // B(NK): n 0..127, 16 halves
  const int kk = (t & 15) * 2, n0 = (t >> 4) * 8;  // B(KN): 2 k-rows x 8 n

  v8f a00 = {0.f,0.f,0.f,0.f,0.f,0.f,0.f,0.f};
  v8f a01 = a00, a10 = a00, a11 = a00;

  for (int k0 = 0; k0 < K; k0 += 32) {
    {  // stage A: one b128 per thread
      const _Float16* src = Ab + (long)(row0 + ar) * lda + k0 + ak;
      __builtin_prefetch((const void*)(src + 32), 0, 1);  // next K slab
#if HAVE_ASYNC_LDS
      async_copy_b128(src, &Ah[ar][ak]);
#else
      *(v8h*)&Ah[ar][ak] = *(const v8h*)src;
#endif
    }
    if (blayKN == 0) {  // stage B from (Nn,K): two b128 per thread
      const _Float16* src = Bb + (long)(col0 + bn) * ldb + k0 + bk;
      __builtin_prefetch((const void*)(src + 32), 0, 1);
#if HAVE_ASYNC_LDS
      async_copy_b128(src, &Bh[bn][bk]);
      async_copy_b128(src + 8, &Bh[bn][bk + 8]);
#else
      *(v8h*)&Bh[bn][bk]     = *(const v8h*)src;
      *(v8h*)&Bh[bn][bk + 8] = *(const v8h*)(src + 8);
#endif
    } else {            // stage B from (K,Nn): transpose, pair-packed b32 stores
      const _Float16* src = Bb + (long)(k0 + kk) * ldb + col0 + n0;
      __builtin_prefetch((const void*)(src + 32L * ldb), 0, 1);
      const v8h r0 = *(const v8h*)src;
      const v8h r1 = *(const v8h*)(src + ldb);
#pragma unroll
      for (int j = 0; j < 8; ++j) {
        v2h pr; pr[0] = r0[j]; pr[1] = r1[j];
        *(v2h*)&Bh[n0 + j][kk] = pr;
      }
    }
#if HAVE_ASYNC_LDS
    wait_async();  // drain this wave's ASYNCcnt before the barrier
#endif
    __syncthreads();
    const v16h fa0 = frag_lds(&Ah[wr][0],      40, 0, lane);
    const v16h fa1 = frag_lds(&Ah[wr + 16][0], 40, 0, lane);
    const v16h fb0 = frag_lds(&Bh[wc][0],      40, 0, lane);
    const v16h fb1 = frag_lds(&Bh[wc + 16][0], 40, 0, lane);
    a00 = WMMA_F16(fa0, fb0, a00);
    a01 = WMMA_F16(fa0, fb1, a01);
    a10 = WMMA_F16(fa1, fb0, a10);
    a11 = WMMA_F16(fa1, fb1, a11);
    __syncthreads();
  }
  store_tile(Cb, Chb, ldc, row0 + wr,      col0 + wc,      M, Nn, a00, alpha, beta, bias, lane);
  store_tile(Cb, Chb, ldc, row0 + wr,      col0 + wc + 16, M, Nn, a01, alpha, beta, bias, lane);
  store_tile(Cb, Chb, ldc, row0 + wr + 16, col0 + wc,      M, Nn, a10, alpha, beta, bias, lane);
  store_tile(Cb, Chb, ldc, row0 + wr + 16, col0 + wc + 16, M, Nn, a11, alpha, beta, bias, lane);
}

// ------------------------------ reductions ---------------------------------
__device__ __forceinline__ float block_sum256(float v, float* sm) {
  const int t = threadIdx.x;
  sm[t] = v; __syncthreads();
  for (int s = 128; s > 0; s >>= 1) { if (t < s) sm[t] += sm[t + s]; __syncthreads(); }
  const float r = sm[0]; __syncthreads();
  return r;
}
__device__ __forceinline__ float block_max256(float v, float* sm) {
  const int t = threadIdx.x;
  sm[t] = v; __syncthreads();
  for (int s = 128; s > 0; s >>= 1) { if (t < s) sm[t] = fmaxf(sm[t], sm[t + s]); __syncthreads(); }
  const float r = sm[0]; __syncthreads();
  return r;
}

// -------------------------- elementwise kernels ----------------------------
// f32 -> f16 copy
__global__ __launch_bounds__(256)
void k_cvt_h(const float* __restrict__ src, _Float16* __restrict__ dst, long total) {
  const long i = (long)blockIdx.x * 256 + threadIdx.x;
  if (i >= total) return;
  dst[i] = (_Float16)src[i];
}

// zero-padded f32 -> f16 copy: dst (drows x dcols) <- src (rows x cols)
__global__ __launch_bounds__(256)
void k_pad_cvt_h(const float* __restrict__ src, _Float16* __restrict__ dst,
                 int rows, int cols, int dcols, long total) {
  const long i = (long)blockIdx.x * 256 + threadIdx.x;
  if (i >= total) return;
  const int r = (int)(i / dcols), c = (int)(i % dcols);
  dst[i] = (r < rows && c < cols) ? (_Float16)src[(long)r * cols + c] : (_Float16)0.f;
}

// xf = concat(mem, x) -> (8, 4096, 256) as f16
__global__ __launch_bounds__(256)
void k_build_xf(const float* __restrict__ x, const float* __restrict__ mem,
                _Float16* __restrict__ xfh, long total) {
  const long i = (long)blockIdx.x * 256 + threadIdx.x;
  if (i >= total) return;
  const long slab = 4096L * 256;
  const long bt = i / slab, rem = i % slab;
  const int b = (int)(bt >> 2), tt = (int)(bt & 3);
  xfh[i] = (_Float16)((tt < 3) ? mem[((long)(b * 3 + tt)) * slab + rem]
                               : x[(long)b * slab + rem]);
}

// zs[row][k] = zc * clip(cos(zc, p1), 0, 1) as f16;  pads zeroed.
__global__ __launch_bounds__(256)
void k_zprep(const float* __restrict__ z, const float* __restrict__ p1,
             _Float16* __restrict__ zsh) {
  __shared__ float sm[256];
  const int t = threadIdx.x;
  const long row = blockIdx.x;
  const int b = (int)(row >> 14);
  const long tn = row & 16383;
  float v = 0.f, p = 0.f;
  if (t < 150) { v = z[((long)b * 150 + t) * 16384 + tn]; p = p1[t]; }
  const float nv = block_sum256(v * v, sm);
  const float np = block_sum256(p * p, sm);
  const float dp = block_sum256(v * p, sm);
  float cs = dp / (fmaxf(sqrtf(nv), 1e-12f) * fmaxf(sqrtf(np), 1e-12f));
  cs = fminf(fmaxf(cs, 0.f), 1.f);
  if (t < 160) zsh[row * 160 + t] = (_Float16)((t < 150) ? v * cs : 0.f);
}

// same but source is cl rows (f32), prompt2
__global__ __launch_bounds__(256)
void k_clprep(const float* __restrict__ clr, const float* __restrict__ p2,
              _Float16* __restrict__ clsh) {
  __shared__ float sm[256];
  const int t = threadIdx.x;
  const long row = blockIdx.x;
  float v = 0.f, p = 0.f;
  if (t < 150) { v = clr[row * 160 + t]; p = p2[t]; }
  const float nv = block_sum256(v * v, sm);
  const float np = block_sum256(p * p, sm);
  const float dp = block_sum256(v * p, sm);
  float cs = dp / (fmaxf(sqrtf(nv), 1e-12f) * fmaxf(sqrtf(np), 1e-12f));
  cs = fminf(fmaxf(cs, 0.f), 1.f);
  if (t < 160) clsh[row * 160 + t] = (_Float16)((t < 150) ? v * cs : 0.f);
}

// cluster_x_z (b,k,tn) and assigned (b,t,k,n) from mix[(b*16384+tn)*160 + k]
__global__ __launch_bounds__(256)
void k_scatter(const float* __restrict__ mix, float* __restrict__ outC,
               float* __restrict__ outA, long total) {
  const long i = (long)blockIdx.x * 256 + threadIdx.x;
  if (i >= total) return;
  const long tn = i % 16384;
  const long bk = i / 16384;
  const int k = (int)(bk % 150), b = (int)(bk / 150);
  const float v = mix[((long)b * 16384 + tn) * 160 + k];
  outC[i] = v;
  const long tt = tn >> 12, n = tn & 4095;
  outA[(((long)b * 4 + tt) * 150 + k) * 4096 + n] = v;
}

// softmax over n of center[b,t,k,:] -> sfah[(bt*160+k)][n] f16; pad rows -> 0
__global__ __launch_bounds__(256)
void k_softcen(const float* __restrict__ mix, _Float16* __restrict__ sfah) {
  __shared__ float sm[256];
  const int k = blockIdx.x, bt = blockIdx.y, t = threadIdx.x;
  _Float16* dst = sfah + ((long)bt * 160 + k) * 4096;
  if (k >= 150) {
#pragma unroll
    for (int i = 0; i < 16; ++i) dst[t + i * 256] = (_Float16)0.f;
    return;
  }
  const float* src = mix + ((long)bt * 4096) * 160 + k;
  float vals[16];
  float m = -3.0e38f;
#pragma unroll
  for (int i = 0; i < 16; ++i) { vals[i] = src[(long)(t + i * 256) * 160]; m = fmaxf(m, vals[i]); }
  m = block_max256(m, sm);
  float s = 0.f;
#pragma unroll
  for (int i = 0; i < 16; ++i) { vals[i] = __expf(vals[i] - m); s += vals[i]; }
  s = block_sum256(s, sm);
  const float inv = 1.f / s;
#pragma unroll
  for (int i = 0; i < 16; ++i) dst[t + i * 256] = (_Float16)(vals[i] * inv);
}

// cosine-gated memory fusion + LayerNorm -> C_in f16 (pad rows zeroed)
__global__ __launch_bounds__(256)
void k_gate_ln(const float* __restrict__ cen, const float* __restrict__ salpha,
               const float* __restrict__ sbeta, const float* __restrict__ nw,
               const float* __restrict__ nb, _Float16* __restrict__ cinh) {
  __shared__ float sm[256];
  const int k = blockIdx.x, b = blockIdx.y, c = threadIdx.x;
  _Float16* dst = cinh + ((long)b * 160 + k) * 256;
  if (k >= 150) { dst[c] = (_Float16)0.f; return; }
  const long ts = 160L * 256;
  const float* base = cen + ((long)b * 4) * ts + (long)k * 256;
  const float last = base[3 * ts + c];
  const float p0 = base[c], p1v = base[ts + c], p2 = base[2 * ts + c];
  const float nl = sqrtf(block_sum256(last * last, sm));
  const float d0 = block_sum256(last * p0, sm);
  const float n0 = sqrtf(block_sum256(p0 * p0, sm));
  const float d1 = block_sum256(last * p1v, sm);
  const float n1 = sqrtf(block_sum256(p1v * p1v, sm));
  const float d2 = block_sum256(last * p2, sm);
  const float n2 = sqrtf(block_sum256(p2 * p2, sm));
  const float al = salpha[0], be = sbeta[0];
  const float g0 = 1.f / (1.f + __expf(-(be + al * (d0 / fmaxf(nl * n0, 1e-8f)))));
  const float g1 = 1.f / (1.f + __expf(-(be + al * (d1 / fmaxf(nl * n1, 1e-8f)))));
  const float g2 = 1.f / (1.f + __expf(-(be + al * (d2 / fmaxf(nl * n2, 1e-8f)))));
  const float v = last + g0 * p0 + g1 * p1v + g2 * p2;
  const float mean = block_sum256(v, sm) * (1.f / 256.f);
  const float dv = v - mean;
  const float var = block_sum256(dv * dv, sm) * (1.f / 256.f);
  dst[c] = (_Float16)(dv * rsqrtf(var + 1e-5f) * nw[c] + nb[c]);
}

// out = res + LayerNorm(val); optional f16 mirror for downstream GEMM A.
__global__ __launch_bounds__(256)
void k_add_ln(const float* __restrict__ res, const float* __restrict__ val,
              const float* __restrict__ nw, const float* __restrict__ nb,
              float* __restrict__ out, _Float16* __restrict__ outh) {
  __shared__ float sm[256];
  const long row = blockIdx.x;
  const int c = threadIdx.x;
  const float v = val[row * 256 + c];
  const float mean = block_sum256(v, sm) * (1.f / 256.f);
  const float dv = v - mean;
  const float var = block_sum256(dv * dv, sm) * (1.f / 256.f);
  const float r = res[row * 256 + c] + dv * rsqrtf(var + 1e-5f) * nw[c] + nb[c];
  out[row * 256 + c] = r;
  if (outh) outh[row * 256 + c] = (_Float16)r;
}

// depthwise 3x3 SAME conv (+bias) then exact GELU -> f16
__global__ __launch_bounds__(256)
void k_dwconv_gelu(const float* __restrict__ h, const float* __restrict__ w,
                   const float* __restrict__ bias, _Float16* __restrict__ hgh, long total) {
  const long i = (long)blockIdx.x * 256 + threadIdx.x;
  if (i >= total) return;
  const int ch = (int)(i % 1024);
  const long rest = i / 1024;
  const int xx = (int)(rest & 63);
  const int yy = (int)((rest >> 6) & 63);
  const int b = (int)(rest >> 12);
  float acc = bias[ch];
#pragma unroll
  for (int dy = 0; dy < 3; ++dy) {
    const int y2 = yy + dy - 1;
    if (y2 < 0 || y2 >= 64) continue;
#pragma unroll
    for (int dx = 0; dx < 3; ++dx) {
      const int x2 = xx + dx - 1;
      if (x2 < 0 || x2 >= 64) continue;
      acc += h[(((long)b * 4096) + y2 * 64 + x2) * 1024 + ch] * w[ch * 9 + dy * 3 + dx];
    }
  }
  hgh[i] = (_Float16)(0.5f * acc * (1.f + erff(acc * 0.70710678118654752f)));
}

// ------------------------- fused attention kernel --------------------------
// Per (b,h), 128 query rows/block (16/wave). hd=32 == WMMA K, so QK^T is one
// WMMA per 16-key tile; softmax in C-fragment registers via 16-lane shfl_xor;
// P staged to LDS; V staged once per block (pair-packed b32); PV = 5 k-steps.
__global__ __launch_bounds__(256)
void k_attn(const _Float16* __restrict__ qh, const _Float16* __restrict__ kh,
            const _Float16* __restrict__ vh, _Float16* __restrict__ obh) {
  __shared__ _Float16 Pl[8][16][160];  // 40 KB
  __shared__ _Float16 Vh[32][168];     // V^T: [d][key]
  const int lane = threadIdx.x & 31;
  const int wv = threadIdx.x >> 5;
  const int b = blockIdx.z, h = blockIdx.y;
  const int qr0 = blockIdx.x * 128 + wv * 16;
  const _Float16* Q  = qh + ((long)b * 4096) * 256 + h * 32;
  const _Float16* Km = kh + (long)b * 160 * 256 + h * 32;
  const _Float16* Vm = vh + (long)b * 160 * 256 + h * 32;

  // stage V transposed into LDS, two keys per b32 store
  for (int e = threadIdx.x; e < 80 * 32; e += 256) {
    const int key = (e >> 5) * 2, d = e & 31;
    v2h pr;
    pr[0] = Vm[key * 256 + d];
    pr[1] = Vm[(key + 1) * 256 + d];
    *(v2h*)&Vh[d][key] = pr;
  }

  const v16h aq = frag_g16h(Q, 256, qr0, 0, lane);

  v8f logit[10];
#pragma unroll
  for (int f = 0; f < 10; ++f) {
    const v16h bk = frag_g16h(Km, 256, f * 16, 0, lane);  // NK layout == A-pattern
    v8f zc = {0.f,0.f,0.f,0.f,0.f,0.f,0.f,0.f};
    logit[f] = WMMA_F16(aq, bk, zc);
  }
  {  // mask padded keys 150..159 (all in frag 9)
    const int n = 144 + (lane & 15);
    if (n >= 150) {
#pragma unroll
      for (int j = 0; j < 8; ++j) logit[9][j] = -1e30f;
    }
  }
  float mrow[8], srow[8];
#pragma unroll
  for (int j = 0; j < 8; ++j) {
    float m = logit[0][j];
#pragma unroll
    for (int f = 1; f < 10; ++f) m = fmaxf(m, logit[f][j]);
#pragma unroll
    for (int msk = 1; msk < 16; msk <<= 1) m = fmaxf(m, __shfl_xor(m, msk));
    mrow[j] = m;
  }
#pragma unroll
  for (int j = 0; j < 8; ++j) {
    float s = 0.f;
#pragma unroll
    for (int f = 0; f < 10; ++f) {
      const float e = __expf(logit[f][j] - mrow[j]);
      logit[f][j] = e;
      s += e;
    }
#pragma unroll
    for (int msk = 1; msk < 16; msk <<= 1) s += __shfl_xor(s, msk);
    srow[j] = 1.f / s;
  }
  const int nlo = lane & 15;
  const int rh = (lane >> 4) ? 8 : 0;
#pragma unroll
  for (int f = 0; f < 10; ++f)
#pragma unroll
    for (int j = 0; j < 8; ++j)
      Pl[wv][rh + j][f * 16 + nlo] = (_Float16)(logit[f][j] * srow[j]);
  __syncthreads();  // covers V staging + P writes

  v8f o0 = {0.f,0.f,0.f,0.f,0.f,0.f,0.f,0.f};
  v8f o1 = o0;
#pragma unroll
  for (int ks = 0; ks < 5; ++ks) {
    const v16h ap  = frag_lds(&Pl[wv][0][0], 160, ks * 32, lane);
    const v16h bv0 = frag_lds(&Vh[0][0],  168, ks * 32, lane);
    const v16h bv1 = frag_lds(&Vh[16][0], 168, ks * 32, lane);
    o0 = WMMA_F16(ap, bv0, o0);
    o1 = WMMA_F16(ap, bv1, o1);
  }
#pragma unroll
  for (int j = 0; j < 8; ++j) {
    const long r = (long)b * 4096 + qr0 + rh + j;
    obh[r * 256 + h * 32 + nlo]      = (_Float16)o0[j];
    obh[r * 256 + h * 32 + 16 + nlo] = (_Float16)o1[j];
  }
}

// ---------------------------------------------------------------------------
extern "C" void kernel_launch(void* const* d_in, const int* in_sizes, int n_in,
                              void* d_out, int out_size, void* d_ws, size_t ws_size,
                              hipStream_t stream) {
  const float* x      = (const float*)d_in[0];
  const float* z      = (const float*)d_in[1];
  const float* mem    = (const float*)d_in[2];
  const float* cw     = (const float*)d_in[3];
  const float* p1     = (const float*)d_in[4];
  const float* tdt1   = (const float*)d_in[5];
  const float* p2     = (const float*)d_in[6];
  const float* tdt2   = (const float*)d_in[7];
  const float* salpha = (const float*)d_in[8];
  const float* sbeta  = (const float*)d_in[9];
  const float* q_w = (const float*)d_in[10]; const float* q_b = (const float*)d_in[11];
  const float* k_w = (const float*)d_in[12]; const float* k_b = (const float*)d_in[13];
  const float* v_w = (const float*)d_in[14]; const float* v_b = (const float*)d_in[15];
  const float* pr_w = (const float*)d_in[16]; const float* pr_b = (const float*)d_in[17];
  const float* nw = (const float*)d_in[18]; const float* nb = (const float*)d_in[19];
  const float* fc1_w = (const float*)d_in[20]; const float* fc1_b = (const float*)d_in[21];
  const float* dw_w = (const float*)d_in[22]; const float* dw_b = (const float*)d_in[23];
  const float* fc2_w = (const float*)d_in[24]; const float* fc2_b = (const float*)d_in[25];

  // ---- f32 workspace ----
  float* W = (float*)d_ws;
  float* clr  = W; W += 5242880;   // (32768,160)
  float* mix  = W; W += 5242880;
  float* cen  = W; W += 327680;    // (8,160,256)
  float* qbuf = W; W += 2097152;   // f32 mirror of q (unused downstream)
  float* kmat = W; W += 81920;
  float* vmat = W; W += 81920;
  float* o2   = W; W += 2097152;
  float* oat  = W; W += 2097152;
  float* hb   = W; W += 8388608;   // (8192,1024)
  float* h2   = W; W += 2097152;
  // ---- f16 workspace ----
  _Float16* H = (_Float16*)W;
  _Float16* xh     = H; H += 2097152;   // x as f16
  _Float16* xfh    = H; H += 8388608;   // concat(mem,x)
  _Float16* cwph   = H; H += 65536;     // clustering_w padded 256x256
  _Float16* tdt1ph = H; H += 40960;     // tdt1 padded 256x160
  _Float16* tdt2ph = H; H += 40960;
  _Float16* zsh    = H; H += 5242880;
  _Float16* clsh   = H; H += 5242880;
  _Float16* sfah   = H; H += 5242880;
  _Float16* cinh   = H; H += 81920;
  _Float16* qwh    = H; H += 65536;
  _Float16* kwh    = H; H += 65536;
  _Float16* vwh    = H; H += 65536;
  _Float16* prwh   = H; H += 65536;
  _Float16* fc1wh  = H; H += 262144;
  _Float16* fc2wh  = H; H += 262144;
  _Float16* qhh    = H; H += 2097152;
  _Float16* khh    = H; H += 81920;
  _Float16* vhh    = H; H += 81920;
  _Float16* obh    = H; H += 2097152;
  _Float16* oath   = H; H += 2097152;
  _Float16* hgh    = H; H += 8388608;

  float* out0 = (float*)d_out;      // (2,4096,256)
  float* outC = out0 + 2097152;     // cluster_x_z (2,150,16384)
  float* outA = outC + 4915200;     // assigned (2,4,150,4096)

  auto gemm = [&](const _Float16* A, long sA, int lda, const _Float16* Bm, long sB, int ldb,
                  float* Cm, _Float16* Chm, long sC, int ldc, int M, int Nn, int K, int blayKN,
                  float alpha, float beta, const float* bias, int batch) {
    dim3 g((Nn + 127) / 128, (M + 63) / 64, batch);
    gemm_wmma_f16<<<g, 256, 0, stream>>>(A, sA, lda, Bm, sB, ldb, Cm, Chm, sC, ldc,
                                         M, Nn, K, blayKN, alpha, beta, bias);
  };
  auto cvt = [&](const float* s, _Float16* d, long n) {
    k_cvt_h<<<(unsigned)((n + 255) / 256), 256, 0, stream>>>(s, d, n);
  };

  // 0) one-time f32 -> f16 operand conversions (halves GEMM HBM traffic)
  cvt(x, xh, 2097152L);
  cvt(q_w, qwh, 65536L);  cvt(k_w, kwh, 65536L);  cvt(v_w, vwh, 65536L);
  cvt(pr_w, prwh, 65536L);
  cvt(fc1_w, fc1wh, 262144L);  cvt(fc2_w, fc2wh, 262144L);
  k_pad_cvt_h<<<(65536 + 255) / 256, 256, 0, stream>>>(cw,   cwph,   150, 256, 256, 65536L);
  k_pad_cvt_h<<<(40960 + 255) / 256, 256, 0, stream>>>(tdt1, tdt1ph, 150, 150, 160, 40960L);
  k_pad_cvt_h<<<(40960 + 255) / 256, 256, 0, stream>>>(tdt2, tdt2ph, 150, 150, 160, 40960L);
  // 1) xf = concat(mem, x)  (f16)
  k_build_xf<<<32768, 256, 0, stream>>>(x, mem, xfh, 8388608L);
  // 2) cl = xf @ Wc^T
  gemm(xfh, 0, 256, cwph, 0, 256, clr, nullptr, 0, 160, 32768, 160, 256, 0,
       1.f, 0.f, nullptr, 1);
  // 3) prompt-cosine row scaling (write f16 GEMM operands)
  k_zprep <<<32768, 256, 0, stream>>>(z, p1, zsh);
  k_clprep<<<32768, 256, 0, stream>>>(clr, p2, clsh);
  // 4) mix = 0.5*(zs@tdt1) + 0.5*(cls@tdt2)
  gemm(zsh,  0, 160, tdt1ph, 0, 160, mix, nullptr, 0, 160, 32768, 160, 160, 1,
       0.5f, 0.f, nullptr, 1);
  gemm(clsh, 0, 160, tdt2ph, 0, 160, mix, nullptr, 0, 160, 32768, 160, 160, 1,
       0.5f, 1.f, nullptr, 1);
  // 5) emit cluster_x_z + assigned_results
  k_scatter<<<(4915200 + 255) / 256, 256, 0, stream>>>(mix, outC, outA, 4915200L);
  // 6) softmax over N per (b,t,k) -> f16
  k_softcen<<<dim3(160, 8), 256, 0, stream>>>(mix, sfah);
  // 7) cen = soft @ xf  batched over 8 (b,t)
  gemm(sfah, 160L * 4096, 4096, xfh, 4096L * 256, 256, cen, nullptr, 160L * 256, 256,
       160, 256, 4096, 1, 1.f, 0.f, nullptr, 8);
  // 8) gated memory fusion + LN -> C_in (f16)
  k_gate_ln<<<dim3(160, 2), 256, 0, stream>>>(cen, salpha, sbeta, nw, nb, cinh);
  // 9) q/k/v projections (f16 mirrors feed attention)
  gemm(xh, 0, 256, qwh, 0, 256, qbuf, qhh, 0, 256, 8192, 256, 256, 0,
       0.17677669529663687f, 0.f, q_b, 1);
  gemm(cinh, 160L * 256, 256, kwh, 0, 256, kmat, khh, 160L * 256, 256, 160, 256, 256, 0,
       1.f, 0.f, k_b, 2);
  gemm(cinh, 160L * 256, 256, vwh, 0, 256, vmat, vhh, 160L * 256, 256, 160, 256, 256, 0,
       1.f, 0.f, v_b, 2);
  // 10) fused attention
  k_attn<<<dim3(32, 8, 2), 256, 0, stream>>>(qhh, khh, vhh, obh);
  // 11) output projection + residual/LN
  gemm(obh, 0, 256, prwh, 0, 256, o2, nullptr, 0, 256, 8192, 256, 256, 0,
       1.f, 0.f, pr_b, 1);
  k_add_ln<<<8192, 256, 0, stream>>>(x, o2, nw, nb, oat, oath);
  // 12) MLP: fc1 -> dwconv+gelu -> fc2 -> residual/LN
  gemm(oath, 0, 256, fc1wh, 0, 256, hb, nullptr, 0, 1024, 8192, 1024, 256, 0,
       1.f, 0.f, fc1_b, 1);
  k_dwconv_gelu<<<32768, 256, 0, stream>>>(hb, dw_w, dw_b, hgh, 8388608L);
  gemm(hgh, 0, 1024, fc2wh, 0, 1024, h2, nullptr, 0, 256, 8192, 256, 1024, 0,
       1.f, 0.f, fc2_b, 1);
  k_add_ln<<<8192, 256, 0, stream>>>(oat, h2, nw, nb, out0, nullptr);

  (void)in_sizes; (void)n_in; (void)out_size; (void)ws_size;
}
